// TinyRecurrentWorkshop_69054484185493
// MI455X (gfx1250) — matrix operand order repaired and checked by reading, not compiled, not run on hardware
//
#include <hip/hip_runtime.h>
#include <hip/hip_bf16.h>
#include <math.h>

// TinyRecurrentWorkshop fused kernel for MI455X (gfx1250, wave32, WMMA).
// One block = 64 rows of state, resident in LDS (f32) for all 8 passes.
// All three weight matrices resident in LDS as transposed bf16 (~314 KB LDS,
// one workgroup per WGP, 8 waves = 2 per SIMD32).

#define DIM       128
#define HID       256
#define ROWS      64
#define NTHREADS  256   // 8 waves of 32

typedef __attribute__((ext_vector_type(16))) __bf16 v16bf;
typedef __attribute__((ext_vector_type(8)))  float  v8f;
typedef __attribute__((ext_vector_type(2)))  __bf16 bf16x2;

// A-matrix 16x32 bf16 fragment (contiguous k order, same as B): lane l16 holds row M=l16;
// lane-group hi selects K halves. p = &buf[row*stride + kTile*32].
__device__ __forceinline__ v16bf load_a_frag(const __bf16* p, int hi) {
  v16bf f;
  const bf16x2* q0 = (const bf16x2*)(p + hi * 16);       // K = hi*16 + {0..7}
  const bf16x2* q1 = (const bf16x2*)(p + hi * 16 + 8);   // K = hi*16 + 8 + {0..7}
#pragma unroll
  for (int r = 0; r < 4; ++r) {
    bf16x2 a = q0[r];
    f[2 * r]         = a[0];
    f[2 * r + 1]     = a[1];
    bf16x2 b = q1[r];
    f[8 + 2 * r]     = b[0];
    f[8 + 2 * r + 1] = b[1];
  }
  return f;
}

// B-matrix 32x16 bf16 fragment from transposed [N][K] weight storage:
// lane l16 supplies column N=l16; p = &wt[(nT*16 + l16)*strideK + kTile*32 + hi*16].
__device__ __forceinline__ v16bf load_b_frag(const __bf16* p) {
  v16bf f;
  const bf16x2* q = (const bf16x2*)p;
#pragma unroll
  for (int r = 0; r < 8; ++r) {
    bf16x2 a = q[r];
    f[2 * r]     = a[0];
    f[2 * r + 1] = a[1];
  }
  return f;
}

__device__ __forceinline__ v8f zero_v8f() {
  v8f z;
#pragma unroll
  for (int r = 0; r < 8; ++r) z[r] = 0.0f;
  return z;
}

__global__ __launch_bounds__(NTHREADS, 2)   // cap VGPRs so 2 waves/SIMD (all 8 waves) fit
void trw_fused_kernel(const float* __restrict__ x,
                      const float* __restrict__ ln_w, const float* __restrict__ ln_b,
                      const float* __restrict__ w1,   const float* __restrict__ b1,
                      const float* __restrict__ w2,   const float* __restrict__ b2,
                      const float* __restrict__ wg,   const float* __restrict__ bg,
                      const int*   __restrict__ passes_p,
                      float* __restrict__ out) {
  __shared__ __align__(16) float  sS [ROWS * DIM];   // f32 state tile (32 KB)
  __shared__ __align__(16) __bf16 sA [ROWS * DIM];   // LayerNorm out, A of GEMM1 (16 KB)
  __shared__ __align__(16) __bf16 sG [ROWS * HID];   // GeLU out,      A of GEMM2 (32 KB)
  __shared__ __align__(16) __bf16 sC [ROWS * HID];   // [state|refined], A of gate (32 KB)
  __shared__ __align__(16) __bf16 sW1t[HID * DIM];   // w1^T [n][k] (64 KB)
  __shared__ __align__(16) __bf16 sW2t[DIM * HID];   // w2^T [n][k] (64 KB)
  __shared__ __align__(16) __bf16 sWgt[DIM * HID];   // wg^T [n][k] (64 KB)
  __shared__ float sB1[HID];
  __shared__ float sB2[DIM], sBg[DIM], sLnW[DIM], sLnB[DIM];

  const int tid = threadIdx.x;

  // ---- Warm L2 for the shared weight matrices (global_prefetch_b8) ----
  for (int off = tid * 64; off < DIM * HID; off += NTHREADS * 64) {
    __builtin_prefetch(w1 + off, 0, 1);
    __builtin_prefetch(w2 + off, 0, 1);
    __builtin_prefetch(wg + off, 0, 1);
  }

  // ---- One-time: state tile load, vectorized b128 (only large HBM read) ----
  {
    const float4* xg4 = (const float4*)(x + (size_t)blockIdx.x * (ROWS * DIM));
    float4* sS4 = (float4*)sS;
    for (int i = tid; i < (ROWS * DIM) / 4; i += NTHREADS) sS4[i] = xg4[i];
  }

  // ---- One-time: weights -> bf16 transposed in LDS (L2-resident source) ----
  for (int i = tid; i < DIM * HID; i += NTHREADS) {
    const int k1 = i >> 8, n1 = i & 255;   // w1 is [128][256] row-major
    sW1t[n1 * DIM + k1] = (__bf16)w1[i];
    const int k2 = i >> 7, n2 = i & 127;   // w2/wg are [256][128] row-major
    sW2t[n2 * HID + k2] = (__bf16)w2[i];
    sWgt[n2 * HID + k2] = (__bf16)wg[i];
  }
  if (tid < HID) sB1[tid] = b1[tid];
  if (tid < DIM) { sB2[tid] = b2[tid]; sBg[tid] = bg[tid];
                   sLnW[tid] = ln_w[tid]; sLnB[tid] = ln_b[tid]; }
  __syncthreads();

  const int passes = passes_p[0];
  const int wid  = tid >> 5;
  const int lane = tid & 31;
  const int l16  = lane & 15;
  const int hi   = lane >> 4;
  const int mT     = wid & 3;           // M tile 0..3  (64 rows)
  const int nBase1 = (wid >> 2) * 8;    // GEMM1: 8 N-tiles -> 256 cols
  const int nBase2 = (wid >> 2) * 4;    // GEMM2/gate: 4 N-tiles -> 128 cols

  for (int p = 0; p < passes; ++p) {
    // ---- Phase 1: LayerNorm (4 lanes per row, shfl_xor quad reduction) ----
    {
      const int row = tid >> 2, q = tid & 3;
      const float* rp = sS + row * DIM + q * 32;
      float sum = 0.0f, ss = 0.0f;
#pragma unroll
      for (int i = 0; i < 32; ++i) { float v = rp[i]; sum += v; ss += v * v; }
      sum += __shfl_xor(sum, 1, 4); sum += __shfl_xor(sum, 2, 4);
      ss  += __shfl_xor(ss,  1, 4); ss  += __shfl_xor(ss,  2, 4);
      const float mean = sum * (1.0f / DIM);
      const float var  = ss * (1.0f / DIM) - mean * mean;
      const float rstd = rsqrtf(var + 1e-5f);
#pragma unroll
      for (int i = 0; i < 32; ++i) {
        const int col = q * 32 + i;
        const float v = rp[i];
        const float h = (v - mean) * rstd * sLnW[col] + sLnB[col];
        sA[row * DIM + col] = (__bf16)h;
        sC[row * HID + col] = (__bf16)v;   // state half of gate-A
      }
    }
    __syncthreads();

    // ---- Phase 2: H = gelu(h @ w1 + b1), 64x256, K=128 ----
    {
      v8f acc[8];
#pragma unroll
      for (int t = 0; t < 8; ++t) acc[t] = zero_v8f();
#pragma unroll
      for (int k = 0; k < 4; ++k) {
        const v16bf a = load_a_frag(sA + (mT * 16 + l16) * DIM + k * 32, hi);
#pragma unroll
        for (int t = 0; t < 8; ++t) {
          const v16bf b = load_b_frag(sW1t + ((nBase1 + t) * 16 + l16) * DIM + k * 32 + hi * 16);
          acc[t] = __builtin_amdgcn_wmma_f32_16x16x32_bf16(false, a, false, b,
                                                           (short)0, acc[t], false, false);
        }
      }
#pragma unroll
      for (int t = 0; t < 8; ++t) {
        const int col = (nBase1 + t) * 16 + l16;
        const float bias = sB1[col];
#pragma unroll
        for (int r = 0; r < 8; ++r) {
          const int row = mT * 16 + hi * 8 + r;
          const float v = acc[t][r] + bias;
          const float g = 0.5f * v * (1.0f + erff(v * 0.70710678118f)); // exact gelu
          sG[row * HID + col] = (__bf16)g;
        }
      }
    }
    __syncthreads();

    // ---- Phase 3a: refined = H @ w2 + b2, 64x128, K=256 (kept in registers) ----
    v8f aR[4];
#pragma unroll
    for (int t = 0; t < 4; ++t) aR[t] = zero_v8f();
#pragma unroll
    for (int k = 0; k < 8; ++k) {
      const v16bf a = load_a_frag(sG + (mT * 16 + l16) * HID + k * 32, hi);
#pragma unroll
      for (int t = 0; t < 4; ++t) {
        const v16bf b = load_b_frag(sW2t + ((nBase2 + t) * 16 + l16) * HID + k * 32 + hi * 16);
        aR[t] = __builtin_amdgcn_wmma_f32_16x16x32_bf16(false, a, false, b,
                                                        (short)0, aR[t], false, false);
      }
    }
#pragma unroll
    for (int t = 0; t < 4; ++t) {
      const int col = (nBase2 + t) * 16 + l16;
      const float bias = sB2[col];
#pragma unroll
      for (int r = 0; r < 8; ++r) {
        const int row = mT * 16 + hi * 8 + r;
        const float ref = aR[t][r] + bias;
        aR[t][r] = ref;                          // keep f32 refined in regs
        sC[row * HID + DIM + col] = (__bf16)ref; // refined half of gate-A
      }
    }
    __syncthreads();

    // ---- Phase 3b: gate = sigmoid([state|refined] @ wg + bg); state += gate*refined ----
    {
      v8f aG[4];
#pragma unroll
      for (int t = 0; t < 4; ++t) aG[t] = zero_v8f();
#pragma unroll
      for (int k = 0; k < 8; ++k) {
        const v16bf a = load_a_frag(sC + (mT * 16 + l16) * HID + k * 32, hi);
#pragma unroll
        for (int t = 0; t < 4; ++t) {
          const v16bf b = load_b_frag(sWgt + ((nBase2 + t) * 16 + l16) * HID + k * 32 + hi * 16);
          aG[t] = __builtin_amdgcn_wmma_f32_16x16x32_bf16(false, a, false, b,
                                                          (short)0, aG[t], false, false);
        }
      }
#pragma unroll
      for (int t = 0; t < 4; ++t) {
        const int col = (nBase2 + t) * 16 + l16;
        const float bias = sBg[col];
#pragma unroll
        for (int r = 0; r < 8; ++r) {
          const int row = mT * 16 + hi * 8 + r;
          const float gate = 1.0f / (1.0f + __expf(-(aG[t][r] + bias)));
          sS[row * DIM + col] += gate * aR[t][r];   // unique (row,col) per lane
        }
      }
    }
    __syncthreads();
  }

  // ---- Final: write state tile to HBM, vectorized b128 ----
  {
    float4* og4 = (float4*)(out + (size_t)blockIdx.x * (ROWS * DIM));
    const float4* sS4 = (const float4*)sS;
    for (int i = tid; i < (ROWS * DIM) / 4; i += NTHREADS) og4[i] = sS4[i];
  }
}

extern "C" void kernel_launch(void* const* d_in, const int* in_sizes, int n_in,
                              void* d_out, int out_size, void* d_ws, size_t ws_size,
                              hipStream_t stream) {
  (void)n_in; (void)out_size; (void)d_ws; (void)ws_size;
  const float* x    = (const float*)d_in[0];
  const float* ln_w = (const float*)d_in[1];
  const float* ln_b = (const float*)d_in[2];
  const float* w1   = (const float*)d_in[3];
  const float* b1   = (const float*)d_in[4];
  const float* w2   = (const float*)d_in[5];
  const float* b2   = (const float*)d_in[6];
  const float* wg   = (const float*)d_in[7];
  const float* bg   = (const float*)d_in[8];
  const int*   pp   = (const int*)d_in[9];
  float* out = (float*)d_out;

  const int n_rows = in_sizes[0] / DIM;      // 262144
  const int blocks = n_rows / ROWS;          // 4096
  trw_fused_kernel<<<blocks, NTHREADS, 0, stream>>>(x, ln_w, ln_b, w1, b1, w2, b2,
                                                    wg, bg, pp, out);
}
